// SpatialAttention_36386962932272
// MI455X (gfx1250) — matrix operand order repaired and checked by reading, not compiled, not run on hardware
//
#include <hip/hip_runtime.h>
#include <hip/hip_bf16.h>
#include <math.h>

// ---------------- problem constants ----------------
constexpr int BB  = 4;
constexpr int CC  = 256;   // channels
constexpr int CQK = 32;    // q/k channels
constexpr int NN  = 4096;  // tokens (64x64)

// ---------------- WMMA helper types ----------------
typedef __attribute__((ext_vector_type(16))) _Float16 v16h;
typedef __attribute__((ext_vector_type(8)))  _Float16 v8h;
typedef __attribute__((ext_vector_type(8)))  float    v8f;

__device__ __forceinline__ v8f wmma_f16(v16h a, v16h b, v8f c) {
  // D = A(16x32 f16) x B(32x16 f16) + C(16x16 f32)
  return __builtin_amdgcn_wmma_f32_16x16x32_f16(
      /*neg_a=*/false, a, /*neg_b=*/false, b,
      /*c_mod=*/(short)0, c, /*reuse_a=*/false, /*reuse_b=*/false);
}

// A-operand (16x32, f16): lane holds row M = lane&15.
// lanes 0-15 : K = 0..15
// lanes 16-31: K = 16..31   -> k0 = (lane>>4)*16
__device__ __forceinline__ v16h load_a(const _Float16* __restrict__ row, int k0) {
  v8h lo = *(const v8h*)(row + k0);
  v8h hi = *(const v8h*)(row + k0 + 8);
  v16h r;
#pragma unroll
  for (int i = 0; i < 8; ++i) { r[i] = lo[i]; r[i + 8] = hi[i]; }
  return r;
}
// B-operand (32x16, f16): lane holds column N = lane&15, 16 contiguous K,
// K-base = (lane>>4)*16. Memory must be column-major-B == row (N, K contiguous).

// ---------------- kernel 1: x -> Xt (B,N,C) f16 ----------------
__global__ void k_convert_x(const float* __restrict__ x, _Float16* __restrict__ Xt) {
  size_t idx = (size_t)blockIdx.x * blockDim.x + threadIdx.x;
  size_t total = (size_t)BB * CC * NN;
  if (idx >= total) return;
  int n = (int)(idx % NN);
  int c = (int)((idx / NN) % CC);
  int b = (int)(idx / ((size_t)NN * CC));
  Xt[((size_t)b * NN + n) * CC + c] = (_Float16)x[idx];
}

// ---------------- kernel 2: weights -> f16, zero colsum ----------------
__global__ void k_convert_w(const float* __restrict__ wq, const float* __restrict__ wk,
                            const float* __restrict__ wv,
                            _Float16* __restrict__ WqH, _Float16* __restrict__ WkH,
                            _Float16* __restrict__ WvH, float* __restrict__ colsum) {
  int idx = blockIdx.x * blockDim.x + threadIdx.x;
  if (idx < CQK * CC) { WqH[idx] = (_Float16)wq[idx]; WkH[idx] = (_Float16)wk[idx]; }
  if (idx < CC * CC)  { WvH[idx] = (_Float16)wv[idx]; }
  if (idx < BB * NN)  { colsum[idx] = 0.f; }
}

// ---------------- kernel 3: Q,K projection (one wave -> both 16x16 tiles) ---
__global__ void k_proj_qk(const _Float16* __restrict__ Xt,
                          const _Float16* __restrict__ WqH, const _Float16* __restrict__ WkH,
                          const float* __restrict__ bq, const float* __restrict__ bk,
                          _Float16* __restrict__ Qh, _Float16* __restrict__ Kh) {
  int wave = (blockIdx.x * blockDim.x + threadIdx.x) >> 5;
  int lane = threadIdx.x & 31;
  int hl = lane >> 4, col = lane & 15;
  int ot = wave & 1;                 // 2 o-tiles
  int it = (wave >> 1) & 255;        // 256 n-tiles
  int b  = wave >> 9;
  const _Float16* arow = Xt + ((size_t)b * NN + it * 16 + col) * CC;
  int o = ot * 16 + col;
  const _Float16* brq = WqH + (size_t)o * CC + hl * 16;
  const _Float16* brk = WkH + (size_t)o * CC + hl * 16;
  int k0 = hl * 16;
  v8f cq = {}; v8f ck = {};
#pragma unroll
  for (int kk = 0; kk < CC / 32; ++kk) {
    v16h a  = load_a(arow + kk * 32, k0);
    v16h bQ = *(const v16h*)(brq + kk * 32);
    v16h bK = *(const v16h*)(brk + kk * 32);
    cq = wmma_f16(a, bQ, cq);
    ck = wmma_f16(a, bK, ck);
  }
  float biasq = bq[o], biask = bk[o];
#pragma unroll
  for (int v = 0; v < 8; ++v) {
    int row = it * 16 + v + hl * 8;
    Qh[((size_t)b * NN + row) * CQK + o] = (_Float16)(cq[v] + biasq);
    Kh[((size_t)b * NN + row) * CQK + o] = (_Float16)(ck[v] + biask);
  }
}

// ---------------- kernel 4: V projection -> Vh (B,C,N) f16 ----------------
__global__ void k_proj_v(const _Float16* __restrict__ Xt, const _Float16* __restrict__ WvH,
                         const float* __restrict__ bv, _Float16* __restrict__ Vh) {
  int wave = (blockIdx.x * blockDim.x + threadIdx.x) >> 5;
  int lane = threadIdx.x & 31;
  int hl = lane >> 4, col = lane & 15;
  int nt = wave & 255;
  int ct = (wave >> 8) & 15;
  int b  = wave >> 12;
  const _Float16* arow = WvH + (size_t)(ct * 16 + col) * CC;                 // A: wv rows (c)
  const _Float16* brow = Xt + ((size_t)b * NN + nt * 16 + col) * CC + hl * 16; // B: Xt rows (n)
  int k0 = hl * 16;
  v8f acc = {};
#pragma unroll
  for (int kk = 0; kk < CC / 32; ++kk) {
    v16h a = load_a(arow + kk * 32, k0);
    v16h bm = *(const v16h*)(brow + kk * 32);
    acc = wmma_f16(a, bm, acc);
  }
#pragma unroll
  for (int v = 0; v < 8; ++v) {
    int crow = ct * 16 + v + hl * 8;
    Vh[((size_t)b * CC + crow) * NN + nt * 16 + col] = (_Float16)(acc[v] + bv[crow]);
  }
}

// ---------------- kernel 5: online softmax stats (row max / row sum) -------
__global__ void k_stats(const _Float16* __restrict__ Qh, const _Float16* __restrict__ Kh,
                        float* __restrict__ mrow, float* __restrict__ lrow) {
  int wave = (blockIdx.x * blockDim.x + threadIdx.x) >> 5;
  int lane = threadIdx.x & 31;
  int hl = lane >> 4, col = lane & 15;
  int it = wave & 255;
  int b  = wave >> 8;
  int k0 = hl * 16;
  const _Float16* arow = Qh + ((size_t)b * NN + it * 16 + col) * CQK;
  v16h a = load_a(arow, k0);               // A constant for all j-tiles
  float m[8], l[8];
#pragma unroll
  for (int v = 0; v < 8; ++v) { m[v] = -__builtin_inff(); l[v] = 0.f; }
  for (int jt = 0; jt < NN / 16; ++jt) {
    const _Float16* brow = Kh + ((size_t)b * NN + jt * 16 + col) * CQK + hl * 16;
    v16h bm = *(const v16h*)brow;
    v8f z = {};
    v8f s = wmma_f16(a, bm, z);
#pragma unroll
    for (int v = 0; v < 8; ++v) {
      float rm = s[v];
      rm = fmaxf(rm, __shfl_xor(rm, 1, 32));
      rm = fmaxf(rm, __shfl_xor(rm, 2, 32));
      rm = fmaxf(rm, __shfl_xor(rm, 4, 32));
      rm = fmaxf(rm, __shfl_xor(rm, 8, 32));
      float mn = fmaxf(m[v], rm);
      float e = __expf(s[v] - mn);
      e += __shfl_xor(e, 1, 32);
      e += __shfl_xor(e, 2, 32);
      e += __shfl_xor(e, 4, 32);
      e += __shfl_xor(e, 8, 32);
      l[v] = l[v] * __expf(m[v] - mn) + e;
      m[v] = mn;
    }
  }
  if (col == 0) {
#pragma unroll
    for (int v = 0; v < 8; ++v) {
      int row = it * 16 + v + hl * 8;
      mrow[(size_t)b * NN + row] = m[v];
      lrow[(size_t)b * NN + row] = l[v];
    }
  }
}

// ---------------- kernel 6: P = softmax (f16), colsum accumulation ---------
__global__ void k_softmax_p(const _Float16* __restrict__ Qh, const _Float16* __restrict__ Kh,
                            const float* __restrict__ mrow, const float* __restrict__ lrow,
                            _Float16* __restrict__ P, float* __restrict__ colsum) {
  int wave = (blockIdx.x * blockDim.x + threadIdx.x) >> 5;
  int lane = threadIdx.x & 31;
  int hl = lane >> 4, col = lane & 15;
  int jg = wave & 15;                // 16 j-groups of 16 j-tiles
  int it = (wave >> 4) & 255;
  int b  = wave >> 12;
  int k0 = hl * 16;
  const _Float16* arow = Qh + ((size_t)b * NN + it * 16 + col) * CQK;
  v16h a = load_a(arow, k0);
  float mv[8], li[8];
#pragma unroll
  for (int v = 0; v < 8; ++v) {
    int row = it * 16 + v + hl * 8;
    mv[v] = mrow[(size_t)b * NN + row];
    li[v] = 1.0f / lrow[(size_t)b * NN + row];
  }
  for (int jt = jg * 16; jt < jg * 16 + 16; ++jt) {
    const _Float16* brow = Kh + ((size_t)b * NN + jt * 16 + col) * CQK + hl * 16;
    v16h bm = *(const v16h*)brow;
    v8f z = {};
    v8f s = wmma_f16(a, bm, z);
    float cs = 0.f;
#pragma unroll
    for (int v = 0; v < 8; ++v) {
      float p = __expf(s[v] - mv[v]) * li[v];
      cs += p;
      int row = it * 16 + v + hl * 8;
      P[((size_t)b * NN + row) * NN + jt * 16 + col] = (_Float16)p;
    }
    cs += __shfl_xor(cs, 16, 32);    // add both row-halves of this column
    if (hl == 0) atomicAdd(&colsum[(size_t)b * NN + jt * 16 + col], cs);
  }
}

// ---------------- kernel 7: attended = V @ P^T; out = gamma*att + x --------
__global__ void k_attend(const _Float16* __restrict__ Vh, const _Float16* __restrict__ P,
                         const float* __restrict__ x, const float* __restrict__ gamma,
                         float* __restrict__ out) {
  int wave = (blockIdx.x * blockDim.x + threadIdx.x) >> 5;
  int lane = threadIdx.x & 31;
  int hl = lane >> 4, col = lane & 15;
  int it = wave & 255;               // i (query) tile
  int ct = (wave >> 8) & 15;         // c tile
  int b  = wave >> 12;
  int k0 = hl * 16;
  const _Float16* arow = Vh + ((size_t)b * CC + ct * 16 + col) * NN;            // A: V rows (c), K=j
  const _Float16* brow = P + ((size_t)b * NN + it * 16 + col) * NN + hl * 16;   // B: P rows (i), K=j
  v8f acc = {};
#pragma unroll 4
  for (int kk = 0; kk < NN / 32; ++kk) {
    v16h a  = load_a(arow + kk * 32, k0);
    v16h bm = *(const v16h*)(brow + kk * 32);
    acc = wmma_f16(a, bm, acc);
  }
  float g = gamma[0];
#pragma unroll
  for (int v = 0; v < 8; ++v) {
    int crow = ct * 16 + v + hl * 8;
    size_t idx = ((size_t)b * CC + crow) * NN + it * 16 + col;
    out[idx] = g * acc[v] + x[idx];
  }
}

// ---------------- kernel 8: attention map = colsum / N ----------------
__global__ void k_map(const float* __restrict__ colsum, float* __restrict__ amap) {
  int idx = blockIdx.x * blockDim.x + threadIdx.x;
  if (idx < BB * NN) amap[idx] = colsum[idx] * (1.0f / (float)NN);
}

// ---------------- launcher ----------------
extern "C" void kernel_launch(void* const* d_in, const int* in_sizes, int n_in,
                              void* d_out, int out_size, void* d_ws, size_t ws_size,
                              hipStream_t stream) {
  const float* x     = (const float*)d_in[0];
  const float* wq    = (const float*)d_in[1];
  const float* bq    = (const float*)d_in[2];
  const float* wk    = (const float*)d_in[3];
  const float* bk    = (const float*)d_in[4];
  const float* wv    = (const float*)d_in[5];
  const float* bv    = (const float*)d_in[6];
  const float* gamma = (const float*)d_in[7];
  float* out  = (float*)d_out;                       // (B,C,H,W) fp32
  float* amap = out + (size_t)BB * CC * NN;          // (B,H,W)   fp32

  char* w = (char*)d_ws;
  size_t off = 0;
  auto carve = [&](size_t bytes) -> char* {
    char* p = w + off;
    off += (bytes + 255) & ~(size_t)255;
    return p;
  };
  _Float16* Xt  = (_Float16*)carve((size_t)BB * NN * CC * 2);   // 8 MB
  _Float16* WqH = (_Float16*)carve((size_t)CQK * CC * 2);
  _Float16* WkH = (_Float16*)carve((size_t)CQK * CC * 2);
  _Float16* WvH = (_Float16*)carve((size_t)CC * CC * 2);
  _Float16* Qh  = (_Float16*)carve((size_t)BB * NN * CQK * 2);  // 1 MB
  _Float16* Kh  = (_Float16*)carve((size_t)BB * NN * CQK * 2);  // 1 MB
  _Float16* Vh  = (_Float16*)carve((size_t)BB * CC * NN * 2);   // 8 MB
  float*    mrow   = (float*)carve((size_t)BB * NN * 4);
  float*    lrow   = (float*)carve((size_t)BB * NN * 4);
  float*    colsum = (float*)carve((size_t)BB * NN * 4);
  _Float16* P   = (_Float16*)carve((size_t)BB * NN * NN * 2);   // 128 MB (fits in 192MB L2)

  // 1) transpose+convert x
  {
    size_t total = (size_t)BB * CC * NN;
    int blocks = (int)((total + 255) / 256);
    k_convert_x<<<blocks, 256, 0, stream>>>(x, Xt);
  }
  // 2) convert weights + zero colsum
  k_convert_w<<<(CC * CC + 255) / 256, 256, 0, stream>>>(wq, wk, wv, WqH, WkH, WvH, colsum);
  // 3) Q/K projection: waves = B * 256 n-tiles * 2 o-tiles = 2048
  k_proj_qk<<<2048 * 32 / 128, 128, 0, stream>>>(Xt, WqH, WkH, bq, bk, Qh, Kh);
  // 4) V projection: waves = B * 16 c-tiles * 256 n-tiles = 16384
  k_proj_v<<<16384 * 32 / 128, 128, 0, stream>>>(Xt, WvH, bv, Vh);
  // 5) softmax stats: waves = B * 256 i-tiles = 1024
  k_stats<<<1024 * 32 / 128, 128, 0, stream>>>(Qh, Kh, mrow, lrow);
  // 6) P + column sums: waves = B * 256 i-tiles * 16 j-groups = 16384
  k_softmax_p<<<16384 * 32 / 128, 128, 0, stream>>>(Qh, Kh, mrow, lrow, P, colsum);
  // 7) attended + residual epilogue: waves = B * 16 c-tiles * 256 i-tiles = 16384
  k_attend<<<16384 * 32 / 128, 128, 0, stream>>>(Vh, P, x, gamma, out);
  // 8) attention map
  k_map<<<(BB * NN + 255) / 256, 256, 0, stream>>>(colsum, amap);
}
